// QuantLinear_54065048322157
// MI455X (gfx1250) — matrix operand order repaired and checked by reading, not compiled, not run on hardware
//
#include <hip/hip_runtime.h>
#include <hip/hip_bf16.h>

typedef __attribute__((ext_vector_type(16))) _Float16 v16h;
typedef __attribute__((ext_vector_type(8)))  _Float16 v8h;
typedef __attribute__((ext_vector_type(2)))  _Float16 h2;
typedef __attribute__((ext_vector_type(8)))  float    v8f;
typedef __attribute__((ext_vector_type(4)))  float    v4f;
typedef __attribute__((ext_vector_type(8)))  unsigned v8u;

#define MM 16
#define KK 4096
#define NN 11008
#define GSIZE 128
#define NGROUPS 32
#define SPLITK 4
#define KSLICE (KK / SPLITK)       // 1024
#define GPB (NGROUPS / SPLITK)     // 8 groups per block
#define WPB 8
#define THREADS (32 * WPB)

// Pass 1: out[m,n] = bias[n]  (split-K partials are atomically added on top)
__global__ __launch_bounds__(256) void init_out_kernel(
    const float* __restrict__ bias, float* __restrict__ out) {
    const int n = blockIdx.x * 256 + threadIdx.x;   // 11008 = 43*256
    const int m = blockIdx.y;
    out[(size_t)m * NN + n] = bias[n];
}

// Pass 2: one wave = one 16x16 output tile over this block's K-slice.
// Magic dequant: f16 0x6400|n == 1024+q; WMMA consumes raw 1024+q values and
// the per-group correction  y_g = s*P_g - (1024*s + z)*S_g[m]  is applied in
// f32, with S_g[m] = sum of the f16-rounded x over the group (LDS row-sum
// table) so the 1024*s*S term cancels exactly.
// K within each 8-nibble word is permuted as pairs (w, w+4); A is staged into
// LDS with the same permutation, which is legal since WMMA contracts slot p
// of A against slot p of B for any consistent K-permutation.
__global__ __launch_bounds__(THREADS) void qgemm_w4_wmma(
    const float*    __restrict__ x,        // (16, 4096) f32
    const unsigned* __restrict__ qweight,  // (11008, 512) u32, 8 nibbles LSB-first
    const float*    __restrict__ scales,   // (32, 11008)
    const float*    __restrict__ zeros,    // (32, 11008)  (stores zero*scale)
    float*          __restrict__ out)      // (16, 11008), pre-filled with bias
{
    __shared__ _Float16 xlds[MM * KSLICE];   // 32 KB
    __shared__ float    rowsum[GPB * MM];    // 8 groups x 16 rows

    const int tid   = threadIdx.x;
    const int kblk  = blockIdx.y;            // K-split index
    const int kbase = kblk * KSLICE;

    for (int i = tid; i < GPB * MM; i += THREADS) rowsum[i] = 0.0f;
    __syncthreads();

    // ---- Stage x slice -> LDS f16 (permuted pairs) + per-group row sums ----
    h2* xlds2 = (h2*)xlds;
    for (int c = tid; c < (MM * KSLICE) / 8; c += THREADS) {   // 2048 chunks
        const int m  = c >> 7;               // 128 8-chunks per row
        const int k8 = (c & 127) * 8;        // local k
        const float* px = x + m * KK + kbase + k8;
        v4f v0 = *(const v4f*)px;
        v4f v1 = *(const v4f*)(px + 4);
        const int base2 = (m * KSLICE + k8) >> 1;
        float sumf = 0.0f;
#pragma unroll
        for (int w = 0; w < 4; ++w) {
            h2 p;                             // {x[k8+w], x[k8+w+4]}
            p[0] = (_Float16)v0[w];
            p[1] = (_Float16)v1[w];
            xlds2[base2 + w] = p;             // ds_store_b32
            sumf += (float)p[0] + (float)p[1];  // sum of f16-rounded values
        }
        atomicAdd(&rowsum[(k8 >> 7) * MM + m], sumf);   // ds_add_f32
    }
    __syncthreads();

    const int lane  = tid & 31;
    const int wave  = tid >> 5;
    const int khalf = lane >> 4;
    const int n0    = (blockIdx.x * WPB + wave) * 16;
    const int ncol  = n0 + (lane & 15);

    const unsigned* qrow = qweight + (size_t)ncol * (KK / 8) + (kbase >> 3);
    const _Float16* arow = xlds + (lane & 15) * KSLICE + khalf * 16;
    const float*    srow = rowsum + khalf * 8;

    v8f acc = {};

    for (int gl = 0; gl < GPB; ++gl) {
        const int g = kblk * GPB + gl;
        const float s  = scales[g * NN + ncol];
        const float zz = 1024.0f * s + zeros[g * NN + ncol];

        if (gl + 1 < GPB)   // global_prefetch_b8 for next group's words
            __builtin_prefetch(qrow + (((gl + 1) * GSIZE + khalf * 16) >> 3), 0, 3);

        v8f P = {};
#pragma unroll
        for (int t = 0; t < 4; ++t) {
            const int k0 = gl * GSIZE + t * 32;   // local k

            // A fragment (pre-permuted; two ds_load_b128)
            v8h alo = *(const v8h*)(arow + k0);
            v8h ahi = *(const v8h*)(arow + k0 + 8);
            v16h a  = __builtin_shufflevector(alo, ahi,
                        0, 1, 2, 3, 4, 5, 6, 7, 8, 9, 10, 11, 12, 13, 14, 15);

            // B fragment: bit-ops only -> raw magic values 1024+q
            const uint2 qw = *(const uint2*)(qrow + ((k0 + khalf * 16) >> 3));
            v8u bw;
#pragma unroll
            for (int hw = 0; hw < 2; ++hw) {
                const unsigned w32 = hw ? qw.y : qw.x;
#pragma unroll
                for (int j = 0; j < 4; ++j)
                    bw[hw * 4 + j] = ((w32 >> (4 * j)) & 0x000F000Fu) | 0x64006400u;
            }
            v16h b = __builtin_bit_cast(v16h, bw);

            P = __builtin_amdgcn_wmma_f32_16x16x32_f16(
                false, a, false, b, (short)0, P, false, false);
        }

        // per-group correction: acc += s*P - zz*S_g[m]
        const v4f S0 = *(const v4f*)(srow + gl * MM);
        const v4f S1 = *(const v4f*)(srow + gl * MM + 4);
#pragma unroll
        for (int r = 0; r < 8; ++r) {
            const float Sm = (r < 4) ? S0[r] : S1[r - 4];
            acc[r] += s * P[r] - zz * Sm;
        }
    }

    // ---- combine split-K partials: hardware global_atomic_add_f32 ----
#pragma unroll
    for (int r = 0; r < 8; ++r)
        unsafeAtomicAdd(out + (size_t)(r + khalf * 8) * NN + ncol, acc[r]);
}

extern "C" void kernel_launch(void* const* d_in, const int* in_sizes, int n_in,
                              void* d_out, int out_size, void* d_ws, size_t ws_size,
                              hipStream_t stream) {
    const float*    x       = (const float*)d_in[0];
    const unsigned* qweight = (const unsigned*)d_in[1];
    const float*    scales  = (const float*)d_in[2];
    const float*    zeros   = (const float*)d_in[3];
    const float*    bias    = (const float*)d_in[4];
    // d_in[5] = g_idx: unused (groups are uniform: g = k / 128)
    float* out = (float*)d_out;

    dim3 ginit(NN / 256, MM);               // (43, 16)
    hipLaunchKernelGGL(init_out_kernel, ginit, dim3(256), 0, stream, bias, out);

    dim3 grid(NN / (16 * WPB), SPLITK);     // (86, 4) = 344 blocks
    hipLaunchKernelGGL(qgemm_w4_wmma, grid, dim3(THREADS), 0, stream,
                       x, qweight, scales, zeros, out);
}
